// Model_17274358465266
// MI455X (gfx1250) — matrix operand order repaired and checked by reading, not compiled, not run on hardware
//
#include <hip/hip_runtime.h>

typedef __attribute__((ext_vector_type(16))) _Float16 v16h;
typedef __attribute__((ext_vector_type(8)))  _Float16 v8h;
typedef __attribute__((ext_vector_type(4)))  _Float16 v4h;
typedef __attribute__((ext_vector_type(8)))  float    v8f;

#define NPTS  (4096 * 64)
#define NRAYS 4096

// level offsets for the instant-NGP grid (base_res=16, 10 levels, T=2^21, align 8)
__constant__ int kOffs[10] = {0, 4096, 36864, 299008, 2396160,
                              4493312, 6590464, 8687616, 10784768, 12881920};

// ---------------------------------------------------------------------------
// Kernel 1: contraction + hash-grid encode -> f16 feats [N][64] (cols 40..63 = 0)
// ---------------------------------------------------------------------------
__global__ __launch_bounds__(256) void grid_encode_kernel(
    const float* __restrict__ coords, const float* __restrict__ emb,
    _Float16* __restrict__ feats)
{
    int p = blockIdx.x * 256 + threadIdx.x;
    float x = coords[3 * p + 0], y = coords[3 * p + 1], z = coords[3 * p + 2];
    // multinerf contraction
    float nn = sqrtf(x * x + y * y + z * z);
    nn = fmaxf(nn, 1.1920929e-07f);
    if (nn > 1.0f) {
        float s = (2.0f - 1.0f / nn) / nn;
        x *= s; y *= s; z *= s;
    }
    // /bound(2) then map [-1,1] -> [0,1]
    float ux = x * 0.25f + 0.5f, uy = y * 0.25f + 0.5f, uz = z * 0.25f + 0.5f;
    _Float16* fout = feats + (size_t)p * 64;
    #pragma unroll 1
    for (int l = 0; l < 10; ++l) {
        int   res   = 16 << l;
        float scale = (float)(res - 1);
        float px = ux * scale + 0.5f, py = uy * scale + 0.5f, pz = uz * scale + 0.5f;
        float gx = floorf(px), gy = floorf(py), gz = floorf(pz);
        float fx = px - gx, fy = py - gy, fz = pz - gz;
        int ix = (int)gx, iy = (int)gy, iz = (int)gz;
        float a0 = 0.f, a1 = 0.f, a2 = 0.f, a3 = 0.f;
        int base = kOffs[l];
        #pragma unroll
        for (int c = 0; c < 8; ++c) {
            int bx = c & 1, by = (c >> 1) & 1, bz = (c >> 2) & 1;
            int cx = ix + bx, cy = iy + by, cz = iz + bz;
            unsigned idx;
            if (l < 4) {  // dense level
                cx = cx < res - 1 ? cx : res - 1;
                cy = cy < res - 1 ? cy : res - 1;
                cz = cz < res - 1 ? cz : res - 1;
                idx = (unsigned)(cx + cy * res + cz * res * res);
            } else {      // spatial hash
                idx = ((unsigned)cx ^ ((unsigned)cy * 2654435761u)
                                   ^ ((unsigned)cz * 805459861u)) & ((1u << 21) - 1u);
            }
            const float4 ev = *(const float4*)(emb + ((size_t)base + idx) * 4);
            float w = (bx ? fx : 1.0f - fx) * (by ? fy : 1.0f - fy) * (bz ? fz : 1.0f - fz);
            a0 += w * ev.x; a1 += w * ev.y; a2 += w * ev.z; a3 += w * ev.w;
        }
        v4h fv;
        fv.x = (_Float16)a0; fv.y = (_Float16)a1; fv.z = (_Float16)a2; fv.w = (_Float16)a3;
        *(v4h*)(fout + l * 4) = fv;
    }
    v8h zz;
    #pragma unroll
    for (int i = 0; i < 8; ++i) zz[i] = (_Float16)0.0f;
    *(v8h*)(fout + 40) = zz; *(v8h*)(fout + 48) = zz; *(v8h*)(fout + 56) = zz;
}

// ---------------------------------------------------------------------------
// Kernel 2: view-dir posenc -> f16 de [R][32] (27 valid + zero pad)
// ---------------------------------------------------------------------------
__global__ __launch_bounds__(256) void direnc_kernel(
    const float* __restrict__ vd, _Float16* __restrict__ de)
{
    int r = blockIdx.x * 256 + threadIdx.x;
    if (r >= NRAYS) return;
    float d0 = vd[3 * r], d1 = vd[3 * r + 1], d2 = vd[3 * r + 2];
    _Float16* o = de + (size_t)r * 32;
    o[0] = (_Float16)d0; o[1] = (_Float16)d1; o[2] = (_Float16)d2;
    float dd[3] = {d0, d1, d2};
    #pragma unroll
    for (int k = 0; k < 4; ++k) {
        float sc = (float)(1 << k);
        #pragma unroll
        for (int j = 0; j < 3; ++j) {
            o[3 + 6 * k + j]     = (_Float16)sinf(dd[j] * sc);
            o[3 + 6 * k + 3 + j] = (_Float16)cosf(dd[j] * sc);
        }
    }
    #pragma unroll
    for (int i = 27; i < 32; ++i) o[i] = (_Float16)0.0f;
}

// ---------------------------------------------------------------------------
// Kernel 3: pack fp32 weight (fi,fo) row-major -> f16 transposed [fo_pad][kpad]
// ---------------------------------------------------------------------------
__global__ __launch_bounds__(256) void pack_kernel(
    const float* __restrict__ src, _Float16* __restrict__ dst,
    int fi, int fo, int fo_pad, int kpad)
{
    int tid = blockIdx.x * 256 + threadIdx.x;
    if (tid >= fo_pad * kpad) return;
    int o = tid / kpad, k = tid % kpad;
    float v = (o < fo && k < fi) ? src[k * fo + o] : 0.0f;
    dst[tid] = (_Float16)v;
}

// ---------------------------------------------------------------------------
// WMMA helpers (layouts per CDNA5 ISA 7.12.2, wave32)
// ---------------------------------------------------------------------------
__device__ __forceinline__ v16h cat8(v8h lo, v8h hi) {
    return __builtin_shufflevector(lo, hi, 0, 1, 2, 3, 4, 5, 6, 7,
                                           8, 9, 10, 11, 12, 13, 14, 15);
}

// A: 16x32 f16, row-major source [16][pitch]; lane m=L&15, g=L>>4:
//    VGPR0-3 = K(g*8+0..7), VGPR4-7 = K(16+g*8+0..7)
__device__ __forceinline__ v16h load_a(const _Float16* base, int pitch, int kt, int lane) {
    int m = lane & 15, g = lane >> 4;
    const _Float16* p = base + m * pitch + kt * 32 + g * 8;
    v8h lo = *(const v8h*)p;
    v8h hi = *(const v8h*)(p + 16);
    return cat8(lo, hi);
}

// B: 32x16 f16 from transposed weights Wt[out][kpad]; lane n=L&15, g=L>>4:
//    VGPR0-3 = K(g*8+0..7), VGPR4-7 = K(16+g*8+0..7)
__device__ __forceinline__ v16h load_b(const _Float16* wt, int kpad, int nt, int kt, int lane) {
    int n = lane & 15, g = lane >> 4;
    const _Float16* p = wt + (size_t)(nt * 16 + n) * kpad + kt * 32 + g * 8;
    v8h lo = *(const v8h*)p;
    v8h hi = *(const v8h*)(p + 16);
    return cat8(lo, hi);
}

__device__ __forceinline__ v8f bias_c(const float* b, int nt, int lane) {
    float x = b[nt * 16 + (lane & 15)];
    v8f c;
    #pragma unroll
    for (int i = 0; i < 8; ++i) c[i] = x;
    return c;
}

// D f32 -> f16 row-major LDS: element (m,n): lane=(m>=8?16:0)+n, vgpr=m&7
__device__ __forceinline__ void store_d(v8f acc, _Float16* colbase, int pitch, int lane, bool relu) {
    int n = lane & 15, g = lane >> 4;
    #pragma unroll
    for (int v = 0; v < 8; ++v) {
        float x = acc[v];
        if (relu) x = fmaxf(x, 0.0f);
        colbase[(g * 8 + v) * pitch + n] = (_Float16)x;
    }
}

#define WMMA(a, b, c) \
    __builtin_amdgcn_wmma_f32_16x16x32_f16(false, (a), false, (b), (short)0, (c), false, false)

// ---------------------------------------------------------------------------
// Kernel 4: fused MLP. Block = 4 waves; each wave owns one 16-point tile.
// Per-wave LDS: act[16][544] = [ x(0..255) | bott(256..511) | de(512..538) | 0 ],
// plus hbuf[16][64] for the 64-wide hidden layer.
// ---------------------------------------------------------------------------
__global__ __launch_bounds__(128) void mlp_kernel(
    const _Float16* __restrict__ feats, const _Float16* __restrict__ de,
    const _Float16* __restrict__ w1t,  const float* __restrict__ b1,
    const _Float16* __restrict__ w2t,  const float* __restrict__ b2,
    const _Float16* __restrict__ l0t,  const float* __restrict__ l0b,
    const _Float16* __restrict__ l1t,  const float* __restrict__ l1b,
    const _Float16* __restrict__ rgbt, const float* __restrict__ rgbb,
    float* __restrict__ dens, float* __restrict__ rgb)
{
    extern __shared__ _Float16 smem[];
    const int lane = threadIdx.x & 31;
    const int wave = threadIdx.x >> 5;
    const int tile = blockIdx.x * 4 + wave;
    _Float16* act  = smem + (size_t)wave * (16 * 544 + 16 * 64);
    _Float16* hbuf = act + 16 * 544;

    // ---- layer 1: feats[16x64(pad)] @ w1 -> relu -> h[16x64]
    {
        v8f acc[4];
        #pragma unroll
        for (int nt = 0; nt < 4; ++nt) acc[nt] = bias_c(b1, nt, lane);
        const _Float16* fb = feats + (size_t)tile * 16 * 64;
        #pragma unroll
        for (int kt = 0; kt < 2; ++kt) {
            v16h a = load_a(fb, 64, kt, lane);
            #pragma unroll
            for (int nt = 0; nt < 4; ++nt) {
                v16h b = load_b(w1t, 64, nt, kt, lane);
                acc[nt] = WMMA(a, b, acc[nt]);
            }
        }
        #pragma unroll
        for (int nt = 0; nt < 4; ++nt) store_d(acc[nt], hbuf + nt * 16, 64, lane, true);
    }
    __syncthreads();

    // ---- layer 2: h @ w2 -> bott[16x256]; density = softplus(bott[:,0]-1)
    {
        v8f acc[16];
        #pragma unroll
        for (int nt = 0; nt < 16; ++nt) acc[nt] = bias_c(b2, nt, lane);
        #pragma unroll
        for (int kt = 0; kt < 2; ++kt) {
            v16h a = load_a(hbuf, 64, kt, lane);
            #pragma unroll
            for (int nt = 0; nt < 16; ++nt) {
                v16h b = load_b(w2t, 64, nt, kt, lane);
                acc[nt] = WMMA(a, b, acc[nt]);
            }
        }
        if ((lane & 15) == 0) {  // column 0 lives in lanes 0 and 16
            #pragma unroll
            for (int v = 0; v < 8; ++v) {
                float raw = acc[0][v] - 1.0f;
                float sp  = (raw > 15.0f) ? raw : log1pf(expf(raw));
                dens[tile * 16 + (lane >> 4) * 8 + v] = sp;
            }
        }
        #pragma unroll
        for (int nt = 0; nt < 16; ++nt)
            store_d(acc[nt], act + 256 + nt * 16, 544, lane, false);
        // broadcast view-dir encoding (whole tile shares one ray: S=64, tiles of 16)
        {
            int r = (tile * 16) >> 6;
            int row = lane & 15, g = lane >> 4;
            const _Float16* s = de + (size_t)r * 32 + g * 16;
            _Float16*       d = act + row * 544 + 512 + g * 16;
            *(v8h*)d       = *(const v8h*)s;
            *(v8h*)(d + 8) = *(const v8h*)(s + 8);
        }
    }
    __syncthreads();

    // ---- lin0: inp[16x283(pad288)] @ lin0_w -> relu -> x in act cols 0..255
    {
        v8f acc[16];
        #pragma unroll
        for (int nt = 0; nt < 16; ++nt) acc[nt] = bias_c(l0b, nt, lane);
        #pragma unroll 1
        for (int kt = 0; kt < 9; ++kt) {
            v16h a = load_a(act + 256, 544, kt, lane);
            #pragma unroll
            for (int nt = 0; nt < 16; ++nt) {
                v16h b = load_b(l0t, 288, nt, kt, lane);
                acc[nt] = WMMA(a, b, acc[nt]);
            }
        }
        #pragma unroll
        for (int nt = 0; nt < 16; ++nt)
            store_d(acc[nt], act + nt * 16, 544, lane, true);
    }
    __syncthreads();

    // ---- lin1: [x|bott|de][16x539(pad544)] @ lin1_w -> relu -> x (overwrite cols 0..255)
    {
        v8f acc[16];
        #pragma unroll
        for (int nt = 0; nt < 16; ++nt) acc[nt] = bias_c(l1b, nt, lane);
        #pragma unroll 1
        for (int kt = 0; kt < 17; ++kt) {
            v16h a = load_a(act, 544, kt, lane);
            #pragma unroll
            for (int nt = 0; nt < 16; ++nt) {
                v16h b = load_b(l1t, 544, nt, kt, lane);
                acc[nt] = WMMA(a, b, acc[nt]);
            }
        }
        #pragma unroll
        for (int nt = 0; nt < 16; ++nt)
            store_d(acc[nt], act + nt * 16, 544, lane, true);
    }
    __syncthreads();

    // ---- rgb: x[16x256] @ rgb_w(pad 16 out) -> sigmoid -> pad-scaled rgb
    {
        float bv = ((lane & 15) < 3) ? rgbb[lane & 15] : 0.0f;
        v8f acc;
        #pragma unroll
        for (int i = 0; i < 8; ++i) acc[i] = bv;
        #pragma unroll 1
        for (int kt = 0; kt < 8; ++kt) {
            v16h a = load_a(act, 544, kt, lane);
            v16h b = load_b(rgbt, 256, 0, kt, lane);
            acc = WMMA(a, b, acc);
        }
        int n = lane & 15;
        if (n < 3) {
            #pragma unroll
            for (int v = 0; v < 8; ++v) {
                int m = (lane >> 4) * 8 + v;
                float s = 1.0f / (1.0f + expf(-acc[v]));
                s = s * 1.002f - 0.001f;
                rgb[(size_t)(tile * 16 + m) * 3 + n] = s;
            }
        }
    }
}

// ---------------------------------------------------------------------------
extern "C" void kernel_launch(void* const* d_in, const int* in_sizes, int n_in,
                              void* d_out, int out_size, void* d_ws, size_t ws_size,
                              hipStream_t stream)
{
    (void)in_sizes; (void)n_in; (void)out_size; (void)ws_size;
    const float* coords   = (const float*)d_in[0];
    // d_in[1] = radius (unused by the reference)
    const float* viewdirs = (const float*)d_in[2];
    const float* emb      = (const float*)d_in[3];
    const float* w1  = (const float*)d_in[4];
    const float* b1  = (const float*)d_in[5];
    const float* w2  = (const float*)d_in[6];
    const float* b2  = (const float*)d_in[7];
    const float* l0w = (const float*)d_in[8];
    const float* l0b = (const float*)d_in[9];
    const float* l1w = (const float*)d_in[10];
    const float* l1b = (const float*)d_in[11];
    const float* rgbw = (const float*)d_in[12];
    const float* rgbb = (const float*)d_in[13];

    char* ws = (char*)d_ws;
    size_t off = 0;
    auto carve = [&](size_t bytes) -> char* {
        char* p = ws + off;
        off += (bytes + 255) & ~(size_t)255;
        return p;
    };
    _Float16* feats = (_Float16*)carve((size_t)NPTS * 64 * 2);
    _Float16* de    = (_Float16*)carve((size_t)NRAYS * 32 * 2);
    _Float16* w1t   = (_Float16*)carve(64 * 64 * 2);
    _Float16* w2t   = (_Float16*)carve(256 * 64 * 2);
    _Float16* l0t   = (_Float16*)carve(256 * 288 * 2);
    _Float16* l1t   = (_Float16*)carve(256 * 544 * 2);
    _Float16* rgbt  = (_Float16*)carve(16 * 256 * 2);

    pack_kernel<<<(64 * 64 + 255) / 256, 256, 0, stream>>>(w1, w1t, 40, 64, 64, 64);
    pack_kernel<<<(256 * 64 + 255) / 256, 256, 0, stream>>>(w2, w2t, 64, 256, 256, 64);
    pack_kernel<<<(256 * 288 + 255) / 256, 256, 0, stream>>>(l0w, l0t, 283, 256, 256, 288);
    pack_kernel<<<(256 * 544 + 255) / 256, 256, 0, stream>>>(l1w, l1t, 539, 256, 256, 544);
    pack_kernel<<<(16 * 256 + 255) / 256, 256, 0, stream>>>(rgbw, rgbt, 256, 3, 16, 256);
    direnc_kernel<<<(NRAYS + 255) / 256, 256, 0, stream>>>(viewdirs, de);
    grid_encode_kernel<<<NPTS / 256, 256, 0, stream>>>(coords, emb, feats);

    float* dens = (float*)d_out;
    float* rgb  = (float*)d_out + NPTS;
    const int shmem = 4 * (16 * 544 + 16 * 64) * 2;  // 77,824 B (<320 KB/WGP)
    hipFuncSetAttribute((const void*)mlp_kernel,
                        hipFuncAttributeMaxDynamicSharedMemorySize, shmem);
    mlp_kernel<<<NPTS / 64, 128, shmem, stream>>>(feats, de, w1t, b1, w2t, b2,
                                                  l0t, l0b, l1t, l1b, rgbt, rgbb,
                                                  dens, rgb);
}
